// AgentGraphAttentionLayer_18700287607225
// MI455X (gfx1250) — matrix operand order repaired and checked by reading, not compiled, not run on hardware
//
#include <hip/hip_runtime.h>
#include <math.h>

#define N_NODES 4096
#define F_IN    256
#define H_HEADS 4
#define D_HEAD  64
#define SLOPE   0.2f
#define LN_EPS  1e-5f

typedef __attribute__((ext_vector_type(16))) __bf16 v16bf;
typedef __attribute__((ext_vector_type(8)))  float  v8f;

__device__ __forceinline__ float leaky(float e) { return fmaxf(e, SLOPE * e); }

// ---------------------------------------------------------------------------
// Kernel 1: Wx = x @ W^T  (4096x256 = 4096x256 * 256x256), bf16 WMMA, f32 acc.
// Each wave computes one 16x16 tile of Wx. Also writes a bf16 transposed copy
// WxT[h*64+d][node] used as the WMMA B operand in the aggregation kernel.
// ---------------------------------------------------------------------------
__global__ void __launch_bounds__(256) gat_wx_gemm(
    const float* __restrict__ x, const float* __restrict__ W,
    float* __restrict__ Wx, __bf16* __restrict__ WxT)
{
    const int wave = threadIdx.x >> 5;
    const int lane = threadIdx.x & 31;
    const int m    = lane & 15;
    const int half = lane >> 4;

    const int tile = blockIdx.x * 8 + wave;   // 4096 tiles total
    const int it = tile >> 4;                 // i-tile 0..255
    const int nt = tile & 15;                 // n-tile 0..15
    const int i0 = it * 16;
    const int n0 = nt * 16;

    v8f acc = {0.f, 0.f, 0.f, 0.f, 0.f, 0.f, 0.f, 0.f};

    const float* xrow = x + (size_t)(i0 + m) * F_IN;   // A row (M = lane&15)
    const float* wrow = W + (size_t)(n0 + m) * F_IN;   // B col (N = lane&15)

    for (int kc = 0; kc < F_IN; kc += 32) {
        // A: x[i0+m][kc + half*8 + {0..7}] and x[i0+m][kc + 16 + half*8 + {0..7}]
        const float4* xa = (const float4*)(xrow + kc + half * 8);
        float4 x0 = xa[0], x1 = xa[1];
        const float4* xb = (const float4*)(xrow + kc + 16 + half * 8);
        float4 x2 = xb[0], x3 = xb[1];
        v16bf a;
        a[0]  = (__bf16)x0.x; a[1]  = (__bf16)x0.y; a[2]  = (__bf16)x0.z; a[3]  = (__bf16)x0.w;
        a[4]  = (__bf16)x1.x; a[5]  = (__bf16)x1.y; a[6]  = (__bf16)x1.z; a[7]  = (__bf16)x1.w;
        a[8]  = (__bf16)x2.x; a[9]  = (__bf16)x2.y; a[10] = (__bf16)x2.z; a[11] = (__bf16)x2.w;
        a[12] = (__bf16)x3.x; a[13] = (__bf16)x3.y; a[14] = (__bf16)x3.z; a[15] = (__bf16)x3.w;

        // B(k,n) = W[n][k]: lane holds column n0+m, k = kc + half*8 + {0..7} and kc + 16 + half*8 + {0..7}
        const float4* wb = (const float4*)(wrow + kc + half * 8);
        const float4* wc = (const float4*)(wrow + kc + 16 + half * 8);
        float4 w0 = wb[0], w1 = wb[1], w2 = wc[0], w3 = wc[1];
        v16bf b;
        b[0]  = (__bf16)w0.x; b[1]  = (__bf16)w0.y; b[2]  = (__bf16)w0.z; b[3]  = (__bf16)w0.w;
        b[4]  = (__bf16)w1.x; b[5]  = (__bf16)w1.y; b[6]  = (__bf16)w1.z; b[7]  = (__bf16)w1.w;
        b[8]  = (__bf16)w2.x; b[9]  = (__bf16)w2.y; b[10] = (__bf16)w2.z; b[11] = (__bf16)w2.w;
        b[12] = (__bf16)w3.x; b[13] = (__bf16)w3.y; b[14] = (__bf16)w3.z; b[15] = (__bf16)w3.w;

        acc = __builtin_amdgcn_wmma_f32_16x16x32_bf16(false, a, false, b,
                                                      (short)0, acc, false, false);
    }

    // C layout: VGPR r, lanes 0-15 -> M=r, lanes 16-31 -> M=r+8; N = lane&15.
    #pragma unroll
    for (int r = 0; r < 8; ++r) {
        const int row = i0 + r + 8 * half;
        const int col = n0 + m;
        const float v = acc[r];
        Wx [(size_t)row * F_IN + col]     = v;
        WxT[(size_t)col * N_NODES + row]  = (__bf16)v;
    }
}

// ---------------------------------------------------------------------------
// Kernel 2: s_src[h][i] = sum_d Wx[i][h*64+d]*a[h][d]; s_dst uses a[h][64+d].
// One block per node row.
// ---------------------------------------------------------------------------
__global__ void __launch_bounds__(256) gat_scores(
    const float* __restrict__ Wx, const float* __restrict__ a,
    float* __restrict__ s_src, float* __restrict__ s_dst)
{
    __shared__ float rs[256];
    __shared__ float rd[256];
    const int i = blockIdx.x;
    const int t = threadIdx.x;
    const int h = t >> 6;
    const int d = t & 63;

    const float v = Wx[(size_t)i * F_IN + t];
    rs[t] = v * a[h * 128 + d];
    rd[t] = v * a[h * 128 + 64 + d];
    __syncthreads();
    #pragma unroll
    for (int s = 32; s > 0; s >>= 1) {
        if (d < s) { rs[t] += rs[t + s]; rd[t] += rd[t + s]; }
        __syncthreads();
    }
    if (d == 0) {
        s_src[h * N_NODES + i] = rs[t];
        s_dst[h * N_NODES + i] = rd[t];
    }
}

// ---------------------------------------------------------------------------
// Kernel 3: per (i,h) masked-softmax stats: c = m + ln(sum exp(e - m)),
// e = LeakyReLU(s_src[i]+s_dst[j]) over adjacent j. One block per row i.
// ---------------------------------------------------------------------------
__global__ void __launch_bounds__(256) gat_softmax_stats(
    const unsigned char* __restrict__ adj,
    const float* __restrict__ s_src, const float* __restrict__ s_dst,
    float* __restrict__ c)
{
    __shared__ float red[H_HEADS][256];
    const int i = blockIdx.x;
    const int t = threadIdx.x;
    const unsigned char* adjRow = adj + (size_t)i * N_NODES;

    float si[H_HEADS];
    #pragma unroll
    for (int h = 0; h < H_HEADS; ++h) si[h] = s_src[h * N_NODES + i];

    float mx[H_HEADS] = {-INFINITY, -INFINITY, -INFINITY, -INFINITY};
    for (int j = t; j < N_NODES; j += 256) {
        if (adjRow[j]) {
            #pragma unroll
            for (int h = 0; h < H_HEADS; ++h) {
                const float e = leaky(si[h] + s_dst[h * N_NODES + j]);
                mx[h] = fmaxf(mx[h], e);
            }
        }
    }
    #pragma unroll
    for (int h = 0; h < H_HEADS; ++h) red[h][t] = mx[h];
    __syncthreads();
    for (int s = 128; s > 0; s >>= 1) {
        if (t < s) {
            #pragma unroll
            for (int h = 0; h < H_HEADS; ++h)
                red[h][t] = fmaxf(red[h][t], red[h][t + s]);
        }
        __syncthreads();
    }
    float mh[H_HEADS];
    #pragma unroll
    for (int h = 0; h < H_HEADS; ++h) mh[h] = red[h][0];
    __syncthreads();

    float sm[H_HEADS] = {0.f, 0.f, 0.f, 0.f};
    for (int j = t; j < N_NODES; j += 256) {
        if (adjRow[j]) {
            #pragma unroll
            for (int h = 0; h < H_HEADS; ++h) {
                const float e = leaky(si[h] + s_dst[h * N_NODES + j]);
                sm[h] += __expf(e - mh[h]);
            }
        }
    }
    #pragma unroll
    for (int h = 0; h < H_HEADS; ++h) red[h][t] = sm[h];
    __syncthreads();
    for (int s = 128; s > 0; s >>= 1) {
        if (t < s) {
            #pragma unroll
            for (int h = 0; h < H_HEADS; ++h)
                red[h][t] += red[h][t + s];
        }
        __syncthreads();
    }
    if (t < H_HEADS)
        c[t * N_NODES + i] = mh[t] + __logf(fmaxf(red[t][0], 1e-30f));
}

// ---------------------------------------------------------------------------
// Kernel 4 (heavy): out_pre[i][h*64+:] = sum_j p(i,j,h) * Wx[j][h*64+:].
// Block = 4 waves for one (16-row i-tile, head): each wave owns a quarter of
// the j-range, generates normalized probabilities p = exp(e - c) directly in
// the WMMA A bf16 register layout, streams WxT tiles as B, and accumulates a
// 16x64 tile with 4 v_wmma_f32_16x16x32_bf16 per 32-j chunk. LDS reduce at end.
// ---------------------------------------------------------------------------
__global__ void __launch_bounds__(128) gat_attend(
    const unsigned char* __restrict__ adj,
    const float* __restrict__ s_src, const float* __restrict__ s_dst,
    const float* __restrict__ c, const __bf16* __restrict__ WxT,
    float* __restrict__ out_pre)
{
    __shared__ float red[4][16][64];

    const int i0   = blockIdx.x * 16;
    const int h    = blockIdx.y;
    const int wave = threadIdx.x >> 5;
    const int lane = threadIdx.x & 31;
    const int m    = lane & 15;
    const int half = lane >> 4;
    const int i    = i0 + m;

    const float si = s_src[h * N_NODES + i];
    const float ci = c    [h * N_NODES + i];
    const unsigned char* adjRow = adj + (size_t)i * N_NODES;
    const float*  sdh   = s_dst + h * N_NODES;
    const __bf16* bbase = WxT + (size_t)(h * D_HEAD) * N_NODES;

    v8f acc[4];
    #pragma unroll
    for (int nt = 0; nt < 4; ++nt)
        acc[nt] = (v8f){0.f, 0.f, 0.f, 0.f, 0.f, 0.f, 0.f, 0.f};

    const int jcBeg = wave * 32, jcEnd = jcBeg + 32;   // 32 chunks of 32 j's
    for (int jc = jcBeg; jc < jcEnd; ++jc) {
        const int j0 = jc * 32;

        // Prefetch next chunk's adjacency + scores (gfx1250 global_prefetch_b8)
        __builtin_prefetch(adjRow + j0 + 32, 0, 1);
        __builtin_prefetch(sdh + j0 + 32, 0, 1);

        // adjacency bits + dst scores for this lane's two k-runs
        const unsigned long long ab0 =
            *(const unsigned long long*)(adjRow + j0 + half * 16);
        const unsigned long long ab1 =
            *(const unsigned long long*)(adjRow + j0 + 8 + half * 16);
        const float4* s0p = (const float4*)(sdh + j0 + half * 16);
        const float4  s00 = s0p[0], s01 = s0p[1];
        const float4* s1p = (const float4*)(sdh + j0 + 8 + half * 16);
        const float4  s10 = s1p[0], s11 = s1p[1];

        v16bf a;
        {
            const float sv[8] = {s00.x, s00.y, s00.z, s00.w,
                                 s01.x, s01.y, s01.z, s01.w};
            #pragma unroll
            for (int tt = 0; tt < 8; ++tt) {
                const float e = leaky(si + sv[tt]);
                const float p = ((ab0 >> (8 * tt)) & 1ull) ? __expf(e - ci) : 0.0f;
                a[tt] = (__bf16)p;
            }
        }
        {
            const float sv[8] = {s10.x, s10.y, s10.z, s10.w,
                                 s11.x, s11.y, s11.z, s11.w};
            #pragma unroll
            for (int tt = 0; tt < 8; ++tt) {
                const float e = leaky(si + sv[tt]);
                const float p = ((ab1 >> (8 * tt)) & 1ull) ? __expf(e - ci) : 0.0f;
                a[8 + tt] = (__bf16)p;
            }
        }

        // B tiles: lane holds column d = nt*16 + m, k-half selects j sub-run
        const __bf16* bj = bbase + j0 + half * 16;
        #pragma unroll
        for (int nt = 0; nt < 4; ++nt) {
            const v16bf b = *(const v16bf*)(bj + (size_t)(nt * 16 + m) * N_NODES);
            acc[nt] = __builtin_amdgcn_wmma_f32_16x16x32_bf16(
                false, a, false, b, (short)0, acc[nt], false, false);
        }
    }

    // Cross-wave reduction of the 16x64 tile through LDS.
    #pragma unroll
    for (int nt = 0; nt < 4; ++nt)
        #pragma unroll
        for (int r = 0; r < 8; ++r)
            red[wave][r + 8 * half][nt * 16 + m] = acc[nt][r];
    __syncthreads();

    for (int e = threadIdx.x; e < 16 * 64; e += 128) {
        const int row = e >> 6;
        const int col = e & 63;
        const float s = red[0][row][col] + red[1][row][col] +
                        red[2][row][col] + red[3][row][col];
        out_pre[(size_t)(i0 + row) * F_IN + h * D_HEAD + col] = s;
    }
}

// ---------------------------------------------------------------------------
// Kernel 5: out = LayerNorm(out_pre + x) * gamma + beta. One block per row.
// ---------------------------------------------------------------------------
__global__ void __launch_bounds__(256) gat_residual_ln(
    const float* __restrict__ out_pre, const float* __restrict__ x,
    const float* __restrict__ gamma, const float* __restrict__ beta,
    float* __restrict__ out)
{
    __shared__ float red[256];
    const int i = blockIdx.x;
    const int t = threadIdx.x;

    const float v = out_pre[(size_t)i * F_IN + t] + x[(size_t)i * F_IN + t];
    red[t] = v;
    __syncthreads();
    for (int s = 128; s > 0; s >>= 1) {
        if (t < s) red[t] += red[t + s];
        __syncthreads();
    }
    const float mu = red[0] * (1.0f / F_IN);
    __syncthreads();

    const float dv = v - mu;
    red[t] = dv * dv;
    __syncthreads();
    for (int s = 128; s > 0; s >>= 1) {
        if (t < s) red[t] += red[t + s];
        __syncthreads();
    }
    const float var = red[0] * (1.0f / F_IN);
    const float r   = rsqrtf(var + LN_EPS);
    out[(size_t)i * F_IN + t] = dv * r * gamma[t] + beta[t];
}

// ---------------------------------------------------------------------------
extern "C" void kernel_launch(void* const* d_in, const int* in_sizes, int n_in,
                              void* d_out, int out_size, void* d_ws, size_t ws_size,
                              hipStream_t stream)
{
    const float*         x     = (const float*)d_in[0];
    const unsigned char* adj   = (const unsigned char*)d_in[1];  // jnp bool
    const float*         W     = (const float*)d_in[2];
    const float*         a     = (const float*)d_in[3];
    const float*         gamma = (const float*)d_in[4];
    const float*         beta  = (const float*)d_in[5];
    float*               out   = (float*)d_out;

    // Workspace layout (~10.2 MB):
    char* ws = (char*)d_ws;
    float*  Wx      = (float*)(ws);                                   // 4 MB
    __bf16* WxT     = (__bf16*)(ws + (size_t)4 * 1024 * 1024);        // 2 MB
    float*  s_src   = (float*)(ws + (size_t)6 * 1024 * 1024);         // 64 KB
    float*  s_dst   = (float*)(ws + (size_t)6 * 1024 * 1024 + 65536); // 64 KB
    float*  cc      = (float*)(ws + (size_t)6 * 1024 * 1024 + 131072);// 64 KB
    float*  out_pre = (float*)(ws + (size_t)6 * 1024 * 1024 + 196608);// 4 MB

    gat_wx_gemm      <<<512,      256, 0, stream>>>(x, W, Wx, WxT);
    gat_scores       <<<N_NODES,  256, 0, stream>>>(Wx, a, s_src, s_dst);
    gat_softmax_stats<<<N_NODES,  256, 0, stream>>>(adj, s_src, s_dst, cc);
    dim3 g(N_NODES / 16, H_HEADS);
    gat_attend       <<<g,        128, 0, stream>>>(adj, s_src, s_dst, cc, WxT, out_pre);
    gat_residual_ln  <<<N_NODES,  256, 0, stream>>>(out_pre, x, gamma, beta, out);
}
